// Agent_44873818309021
// MI455X (gfx1250) — matrix operand order repaired and checked by reading, not compiled, not run on hardware
//
#include <hip/hip_runtime.h>
#include <hip/hip_bf16.h>

#define HDIM 1536
#define EDIM 1536
#define LSEQ 99
#define G4   6144          // 4*H
#define NROWS 208          // 99 sent + 99 words + 10 pad (13 tiles of 16)
#define MTILES 13
#define NTILES 384         // 6144/16
#define NBLK  12           // persistent blocks for the recurrence
#define RPB   (G4 / NBLK)  // 512 gate rows per block
#define KHALF (HDIM / 2)   // 768

typedef __attribute__((ext_vector_type(16))) __bf16 v16bf;
typedef __attribute__((ext_vector_type(8)))  float  v8f;
typedef __attribute__((ext_vector_type(4)))  unsigned int uint4v;

union V16U { v16bf v; uint4v u[2]; };

__device__ __forceinline__ unsigned short f2bf(float f) {
  unsigned u = __float_as_uint(f);
  unsigned r = (u + 0x7FFFu + ((u >> 16) & 1u)) >> 16;
  return (unsigned short)r;
}
__device__ __forceinline__ float bf2f_lo(unsigned pp) { return __uint_as_float((pp & 0xFFFFu) << 16); }
__device__ __forceinline__ float bf2f_hi(unsigned pp) { return __uint_as_float(pp & 0xFFFF0000u); }

__device__ __forceinline__ unsigned rotl32(unsigned v, int d) { return (v << d) | (v >> (32 - d)); }

// Threefry-2x32-20, matching JAX's threefry2x32.
__device__ void threefry(unsigned k0, unsigned k1, unsigned x0, unsigned x1,
                         unsigned& o0, unsigned& o1) {
  unsigned ks2 = k0 ^ k1 ^ 0x1BD11BDAu;
  x0 += k0; x1 += k1;
#define TF4(ra,rb,rc,rd) \
  x0+=x1; x1=rotl32(x1,ra); x1^=x0; \
  x0+=x1; x1=rotl32(x1,rb); x1^=x0; \
  x0+=x1; x1=rotl32(x1,rc); x1^=x0; \
  x0+=x1; x1=rotl32(x1,rd); x1^=x0;
  TF4(13,15,26,6);  x0 += k1;  x1 += ks2 + 1u;
  TF4(17,29,16,24); x0 += ks2; x1 += k0  + 2u;
  TF4(13,15,26,6);  x0 += k0;  x1 += k1  + 3u;
  TF4(17,29,16,24); x0 += k1;  x1 += ks2 + 4u;
  TF4(13,15,26,6);  x0 += ks2; x1 += k0  + 5u;
#undef TF4
  o0 = x0; o1 = x1;
}
__device__ __forceinline__ float u01(unsigned b) {
  return __uint_as_float((b >> 9) | 0x3F800000u) - 1.0f;
}

// device-scope arrive-count barrier (monotonic counter, zeroed per launch)
__device__ __forceinline__ void grid_sync(unsigned* cnt, unsigned target) {
  __syncthreads();
  if (threadIdx.x == 0) {
    __threadfence();
    __hip_atomic_fetch_add(cnt, 1u, __ATOMIC_RELEASE, __HIP_MEMORY_SCOPE_AGENT);
    while (__hip_atomic_load(cnt, __ATOMIC_ACQUIRE, __HIP_MEMORY_SCOPE_AGENT) < target) {
      __builtin_amdgcn_s_sleep(2);
    }
    __threadfence();
  }
  __syncthreads();
}

// ---------------- prep kernels ----------------
__global__ void k_cvt_bf16(const float* __restrict__ src, unsigned short* __restrict__ dst, int n) {
  int i = blockIdx.x * blockDim.x + threadIdx.x;
  int stride = gridDim.x * blockDim.x;
  for (; i < n; i += stride) dst[i] = f2bf(src[i]);
}

__global__ void k_build_A(const float* __restrict__ sent, const float* __restrict__ pos,
                          float* __restrict__ words, unsigned short* __restrict__ A) {
  int row = blockIdx.x;
  for (int e = threadIdx.x; e < EDIM; e += blockDim.x) {
    float v;
    if (row < LSEQ) {
      v = sent[row * EDIM + e];
    } else if (row < 2 * LSEQ) {
      int t = row - LSEQ;
      float w = sent[t * EDIM + e] + pos[t * EDIM + e];
      words[t * EDIM + e] = w;
      v = w;
    } else {
      v = 0.f;
    }
    A[row * EDIM + e] = f2bf(v);
  }
}

__global__ void k_targets(const float* __restrict__ sent, const float* __restrict__ pos,
                          const int* __restrict__ masks,
                          float* __restrict__ targets0, float* __restrict__ tposv) {
  int e = blockIdx.x * blockDim.x + threadIdx.x;
  if (e >= EDIM) return;
  float ms = 0.f, ts = 0.f, tp = 0.f;
  for (int t = 0; t < LSEQ; t++) {
    float m = (float)masks[t];
    ms += m;
    ts += m * sent[t * EDIM + e];
    tp += m * pos[t * EDIM + e];
  }
  targets0[e] = ts / ms;
  tposv[e] = tp / ms;
}

// ---------------- WMMA GEMM: Gx[m][n] = sum_k A[m][k] * W_ih[n][k] ----------------
__global__ __launch_bounds__(128)
void k_gemm(const unsigned short* __restrict__ A,   // [208][1536] bf16
            const unsigned short* __restrict__ B,   // W_ih bf16 [6144][1536]
            float* __restrict__ Gx) {               // [198][6144]
  int wave = (blockIdx.x * blockDim.x + threadIdx.x) >> 5;
  int lane = threadIdx.x & 31;
  int mt = wave % MTILES, nt = wave / MTILES;
  if (nt >= NTILES) return;
  int half_ = lane >> 4, lrow = lane & 15;

  const unsigned short* arow = A + (mt * 16 + lrow) * EDIM + half_ * 8;
  const unsigned short* brow = B + (nt * 16 + lrow) * EDIM + half_ * 8;

  v8f c = {};
  for (int k0 = 0; k0 < EDIM; k0 += 32) {
    V16U a, b;
    a.u[0] = *(const uint4v*)(arow + k0);
    a.u[1] = *(const uint4v*)(arow + k0 + 16);
    b.u[0] = *(const uint4v*)(brow + k0);
    b.u[1] = *(const uint4v*)(brow + k0 + 16);
    c = __builtin_amdgcn_wmma_f32_16x16x32_bf16(false, a.v, false, b.v, (short)0, c, false, false);
  }
  int mbase = mt * 16 + half_ * 8;
  int col = nt * 16 + lrow;
#pragma unroll
  for (int r = 0; r < 8; r++) {
    int m = mbase + r;
    if (m < 2 * LSEQ) Gx[m * G4 + col] = c[r];
  }
}

// row-slice matvec: acc = W_hh[j, khalf*768 .. +768) . h[seg], 128-bit weight loads
__device__ __forceinline__ float dot_half(const unsigned short* __restrict__ wr,
                                          const float* __restrict__ hseg) {
  float acc = 0.f;
  for (int k = 0; k < KHALF; k += 8) {
    uint4v w = *(const uint4v*)(wr + k);
    acc += bf2f_lo(w.x) * hseg[k + 0] + bf2f_hi(w.x) * hseg[k + 1]
         + bf2f_lo(w.y) * hseg[k + 2] + bf2f_hi(w.y) * hseg[k + 3]
         + bf2f_lo(w.z) * hseg[k + 4] + bf2f_hi(w.z) * hseg[k + 5]
         + bf2f_lo(w.w) * hseg[k + 6] + bf2f_hi(w.w) * hseg[k + 7];
  }
  return acc;
}

// ---------------- multi-block persistent recurrence ----------------
__global__ __launch_bounds__(1024)
void k_seq(const float* __restrict__ Gx,               // [198][6144]
           const unsigned short* __restrict__ Whh,     // [6144][1536] bf16
           const unsigned short* __restrict__ W1b,     // [256][6144] bf16
           const float* __restrict__ words,            // [99][1536]
           const float* __restrict__ targets0,
           const float* __restrict__ tposv,
           const float* __restrict__ b_ih, const float* __restrict__ b_hh,
           const float* __restrict__ b1,
           const float* __restrict__ W2, const float* __restrict__ b2,
           const float* __restrict__ Wl, const float* __restrict__ bl,
           const int* __restrict__ labels,
           float* __restrict__ gbuf,                   // [2][6144] ping-pong
           unsigned* __restrict__ cnt,                 // barrier counter (zeroed)
           float* __restrict__ out) {
  __shared__ float hloc[HDIM], cloc[HDIM], over[HDIM], targ[HDIM];
  __shared__ float s1p[1024];
  __shared__ float s1[256];
  __shared__ float z2[2];
  __shared__ float zl[3];
  __shared__ int act_s[LSEQ];
  __shared__ int a_cur;

  int tid = threadIdx.x;
  int jbase = blockIdx.x * RPB;
  int jrow = jbase + (tid >> 1);            // 512 rows/block, 2 threads/row
  int khalf = tid & 1;
  const unsigned short* wrow = Whh + (size_t)jrow * HDIM + khalf * KHALF;

  for (int e = tid; e < HDIM; e += 1024) { hloc[e] = 0.f; cloc[e] = 0.f; }
  __syncthreads();

  unsigned bar = 0;

  // ================= pass 1 =================
  for (int t = 0; t < LSEQ; t++) {
    float* gb = gbuf + (t & 1) * G4;
    // distributed matvec for this block's gate rows
    {
      float acc = dot_half(wrow, hloc + khalf * KHALF);
      acc += __shfl_xor(acc, 1);
      if (khalf == 0)
        gb[jrow] = Gx[(size_t)t * G4 + jrow] + b_ih[jrow] + b_hh[jrow] + acc;
    }
    bar += NBLK; grid_sync(cnt, bar);
    // redundant full gate update (identical in every block)
    for (int e = tid; e < HDIM; e += 1024) {
      float iv = 1.f / (1.f + __expf(-gb[e]));
      float fv = 1.f / (1.f + __expf(-gb[HDIM + e]));
      float gv = tanhf(gb[2 * HDIM + e]);
      float ov = 1.f / (1.f + __expf(-gb[3 * HDIM + e]));
      float c2 = fv * cloc[e] + iv * gv;
      cloc[e] = c2;
      hloc[e] = ov * tanhf(c2);
    }
    __syncthreads();
  }
  for (int e = tid; e < HDIM; e += 1024) {
    over[e] = hloc[e];
    targ[e] = targets0[e];
    hloc[e] = 0.f; cloc[e] = 0.f;
  }
  __syncthreads();

  // ================= pass 2 =================
  for (int t = 0; t < LSEQ; t++) {
    for (int e = tid; e < HDIM; e += 1024) targ[e] += tposv[e];
    __syncthreads();

    float* gb = gbuf + ((LSEQ + t) & 1) * G4;
    // distributed matvec (uses current hloc)
    {
      float acc = dot_half(wrow, hloc + khalf * KHALF);
      acc += __shfl_xor(acc, 1);
      if (khalf == 0)
        gb[jrow] = Gx[(size_t)(LSEQ + t) * G4 + jrow] + b_ih[jrow] + b_hh[jrow] + acc;
    }

    // redundant policy net, overlapped with other blocks' matvec
    {
      int j = tid >> 2, part = tid & 3;
      const unsigned short* wr = W1b + (size_t)j * G4 + part * HDIM;
      const float* seg;
      if      (part == 0) seg = &over[0];
      else if (part == 1) seg = &hloc[0];
      else if (part == 2) seg = words + (size_t)t * EDIM;
      else                seg = &targ[0];
      float acc = 0.f;
      for (int k = 0; k < HDIM; k += 8) {
        uint4v w = *(const uint4v*)(wr + k);
        acc += bf2f_lo(w.x) * seg[k + 0] + bf2f_hi(w.x) * seg[k + 1]
             + bf2f_lo(w.y) * seg[k + 2] + bf2f_hi(w.y) * seg[k + 3]
             + bf2f_lo(w.z) * seg[k + 4] + bf2f_hi(w.z) * seg[k + 5]
             + bf2f_lo(w.w) * seg[k + 6] + bf2f_hi(w.w) * seg[k + 7];
      }
      s1p[tid] = acc;
    }
    __syncthreads();
    if (tid < 256) {
      float v = s1p[tid * 4] + s1p[tid * 4 + 1] + s1p[tid * 4 + 2] + s1p[tid * 4 + 3] + b1[tid];
      s1[tid] = 1.f / (1.f + __expf(-v));
    }
    __syncthreads();
    if (tid < 2) {
      float acc = b2[tid];
      for (int j = 0; j < 256; j++) acc += W2[tid * 256 + j] * s1[j];
      z2[tid] = 1.f / (1.f + __expf(-acc));
    }
    __syncthreads();
    if (tid == 0) {
      float mx = fmaxf(z2[0], z2[1]);
      float e0 = __expf(z2[0] - mx), e1 = __expf(z2[1] - mx);
      float p0 = e0 / (e0 + e1), p1 = e1 / (e0 + e1);
      unsigned kt0, kt1; threefry(0u, 42u, (unsigned)t, (unsigned)(LSEQ + t), kt0, kt1);
      unsigned a0, a1, q0, q1, r0, r1;
      threefry(kt0, kt1, 0u, 3u, a0, a1);   // k1
      threefry(kt0, kt1, 1u, 4u, q0, q1);   // k2
      threefry(kt0, kt1, 2u, 5u, r0, r1);   // k3
      unsigned tb0, tb1; threefry(a0, a1, 0u, 0u, tb0, tb1);
      float tu = u01(tb0);
      unsigned gb0, gb1; threefry(q0, q1, 0u, 1u, gb0, gb1);
      const float tiny = 1.1754943508222875e-38f;
      float u0 = fmaxf(tiny, u01(gb0) * (1.f - tiny) + tiny);
      float u1 = fmaxf(tiny, u01(gb1) * (1.f - tiny) + tiny);
      float gu0 = -__logf(-__logf(u0));
      float gu1 = -__logf(-__logf(u1));
      int acat = (__logf(p0) + gu0 >= __logf(p1) + gu1) ? 0 : 1;
      unsigned rb0, rb1; threefry(r0, r1, 0u, 0u, rb0, rb1);
      int arnd = (int)(rb0 & 1u);
      int a = (tu > 0.1f) ? acat : arnd;
      a_cur = a; act_s[t] = a;
    }

    bar += NBLK; grid_sync(cnt, bar);

    if (a_cur) {  // conditional commit, identical decision in every block
      for (int e = tid; e < HDIM; e += 1024) {
        float iv = 1.f / (1.f + __expf(-gb[e]));
        float fv = 1.f / (1.f + __expf(-gb[HDIM + e]));
        float gv = tanhf(gb[2 * HDIM + e]);
        float ov = 1.f / (1.f + __expf(-gb[3 * HDIM + e]));
        float c2 = fv * cloc[e] + iv * gv;
        cloc[e] = c2;
        hloc[e] = ov * tanhf(c2);
      }
    }
    __syncthreads();
  }

  // ---- classifier + loss: block 0 only ----
  if (blockIdx.x != 0) return;
  if (tid < 3) {
    float acc = bl[tid];
    for (int k = 0; k < HDIM; k++) acc += Wl[tid * HDIM + k] * hloc[k];
    zl[tid] = acc;
  }
  __syncthreads();
  if (tid == 0) {
    float mx = fmaxf(zl[0], fmaxf(zl[1], zl[2]));
    float s = __expf(zl[0] - mx) + __expf(zl[1] - mx) + __expf(zl[2] - mx);
    float lse = mx + __logf(s);
    float p0 = zl[0] - lse, p1 = zl[1] - lse, p2 = zl[2] - lse;
    out[0] = p0; out[1] = p1; out[2] = p2;
    int asum = 0;
    for (int i = 0; i < LSEQ; i++) { out[3 + i] = (float)act_s[i]; asum += act_s[i]; }
    int lab = labels[0];
    float nll = -(lab == 0 ? p0 : (lab == 1 ? p1 : p2));
    float reward = (float)(LSEQ - asum) / (float)LSEQ;
    out[3 + LSEQ] = 2.f * nll * (0.001f + reward);
  }
}

extern "C" void kernel_launch(void* const* d_in, const int* in_sizes, int n_in,
                              void* d_out, int out_size, void* d_ws, size_t ws_size,
                              hipStream_t stream) {
  const float* sents = (const float*)d_in[0];
  const int*   masks = (const int*)d_in[1];
  const int*   labels = (const int*)d_in[2];
  const float* W_ih = (const float*)d_in[3];
  const float* W_hh = (const float*)d_in[4];
  const float* b_ih = (const float*)d_in[5];
  const float* b_hh = (const float*)d_in[6];
  const float* W1   = (const float*)d_in[7];
  const float* b1   = (const float*)d_in[8];
  const float* W2   = (const float*)d_in[9];
  const float* b2   = (const float*)d_in[10];
  const float* Wl   = (const float*)d_in[11];
  const float* bl   = (const float*)d_in[12];
  const float* pos  = (const float*)d_in[13];
  float* out = (float*)d_out;

  char* ws = (char*)d_ws;
  size_t off = 0;
  auto take = [&](size_t bytes) { char* p = ws + off; off = (off + bytes + 255) & ~(size_t)255; return p; };
  unsigned short* Wih_b = (unsigned short*)take((size_t)G4 * EDIM * 2);
  unsigned short* Whh_b = (unsigned short*)take((size_t)G4 * HDIM * 2);
  unsigned short* W1_b  = (unsigned short*)take((size_t)256 * G4 * 2);
  unsigned short* A_b   = (unsigned short*)take((size_t)NROWS * EDIM * 2);
  float* Gx       = (float*)take((size_t)2 * LSEQ * G4 * 4);
  float* wordsBuf = (float*)take((size_t)LSEQ * EDIM * 4);
  float* targets0 = (float*)take((size_t)EDIM * 4);
  float* tposBuf  = (float*)take((size_t)EDIM * 4);
  float* gbuf     = (float*)take((size_t)2 * G4 * 4);
  unsigned* cnt   = (unsigned*)take(256);
  (void)ws_size; (void)in_sizes; (void)n_in; (void)out_size;

  k_cvt_bf16<<<4096, 256, 0, stream>>>(W_ih, Wih_b, G4 * EDIM);
  k_cvt_bf16<<<4096, 256, 0, stream>>>(W_hh, Whh_b, G4 * HDIM);
  k_cvt_bf16<<<2048, 256, 0, stream>>>(W1,   W1_b,  256 * G4);
  k_build_A<<<NROWS, 256, 0, stream>>>(sents, pos, wordsBuf, A_b);
  k_targets<<<(EDIM + 255) / 256, 256, 0, stream>>>(sents, pos, masks, targets0, tposBuf);
  k_gemm<<<(MTILES * NTILES) / 4, 128, 0, stream>>>(A_b, Wih_b, Gx);
  hipMemsetAsync((void*)cnt, 0, sizeof(unsigned), stream);
  k_seq<<<NBLK, 1024, 0, stream>>>(Gx, Whh_b, W1_b, wordsBuf, targets0, tposBuf,
                                   b_ih, b_hh, b1, W2, b2, Wl, bl, labels,
                                   gbuf, cnt, out);
}
